// ClusterModel_55001351193013
// MI455X (gfx1250) — matrix operand order repaired and checked by reading, not compiled, not run on hardware
//
#include <hip/hip_runtime.h>
#include <hip/hip_bf16.h>

// ===========================================================================
// CGCNN-style ClusterModel for MI455X (gfx1250, wave32, WMMA).
//
// Heavy math: per-edge MLP (720000 x 160 @ 160 x 64, then @ 64 x 64) x 3
// blocks -> run through v_wmma_f32_16x16x32_f16 with f16 staged operands.
// Model is HBM-bandwidth bound (~hundreds of MB @ 23.3 TB/s), so gathers are
// fused into the GEMM tile build and weights live as pre-swizzled f16
// fragments (hot in 192MB L2).
// ===========================================================================

typedef _Float16 half_t;
typedef __attribute__((ext_vector_type(16))) _Float16 v16h;
typedef __attribute__((ext_vector_type(8)))  float    v8f;

#define N_ATOMS 60000
#define M_NBR   12
#define NM      (N_ATOMS * M_NBR)      // 720000
#define NBLKS   3
#define B_CRYS  600
#define NPC     100
#define EPSV    1e-5f

#define V8F_ZERO ((v8f){0.f,0.f,0.f,0.f,0.f,0.f,0.f,0.f})

// Swizzled-weight layout offsets (in halfs) within one block's slab
#define E1_OFF 0          // 160x64 = 10240
#define E2_OFF 10240      // 64x64  = 4096
#define V1_OFF 14336
#define V2_OFF 18432
#define U_OFF  22528      // 32x32  = 1024
#define BLK_WSTRIDE 23552

// ---------------- WMMA fragment helpers ----------------
// A matrix 16x32 f16 (row-major tile in LDS, leading dim ldK):
// lane L: row = L%16, half = L/16; VGPR j -> K = half*16 + 2j{+1}.
__device__ inline v16h load_a_frag(const half_t* tile, int ldK, int kbase, int lane) {
  const int row = lane & 15;
  const int hs  = lane >> 4;
  const half_t* rp = tile + row * ldK + kbase + hs * 16;
  v16h f;
#pragma unroll
  for (int j = 0; j < 8; ++j) {
    const int k = j << 1;
    f[2 * j]     = rp[k];
    f[2 * j + 1] = rp[k + 1];
  }
  return f;
}

// B matrix 32x16 f16 fragments pre-swizzled in global memory:
// element e of lane L corresponds to K = (L/16)*16 + e, N = ntile*16 + L%16.
__device__ inline v16h load_b_frag(const half_t* wswz, int chunk, int T, int t, int lane) {
  return *(const v16h*)(wswz + ((size_t)((chunk * T + t) * 32 + lane) << 4));
}

__device__ inline v8f wmma_f16(v16h a, v16h b, v8f c) {
  return __builtin_amdgcn_wmma_f32_16x16x32_f16(false, a, false, b, (short)0, c,
                                                false, false);
}

__device__ inline float bn_val(float x, const float* __restrict__ sum,
                               const float* __restrict__ sumsq,
                               const float* __restrict__ g,
                               const float* __restrict__ b, int feat, float invc) {
  const float mu  = sum[feat] * invc;
  const float var = sumsq[feat] * invc - mu * mu;
  return (x - mu) * rsqrtf(var + EPSV) * g[feat] + b[feat];
}

// ---------------- weight swizzle: f32 row-major (K x Ncols) -> f16 B-fragments
__global__ void swizzle_w_kernel(const float* __restrict__ W, half_t* __restrict__ out,
                                 int K, int Ncols) {
  const int i = blockIdx.x * blockDim.x + threadIdx.x;
  if (i >= K * Ncols) return;
  const int e    = i & 15;
  const int lane = (i >> 4) & 31;
  const int ct   = i >> 9;            // chunk*T + ntile
  const int T    = Ncols >> 4;
  const int t    = ct % T, c = ct / T;
  const int n    = t * 16 + (lane & 15);
  const int k    = c * 32 + ((lane >> 4) << 4) + e;
  out[i] = (half_t)W[k * Ncols + n];
}

// ---------------- generic small embedding GEMM (VALU; tiny share of FLOPs)
__global__ void emb_kernel(const float* __restrict__ X, const float* __restrict__ W,
                           const float* __restrict__ b, float* __restrict__ Y,
                           int rows, int Kin, int Ncols) {
  const int i = blockIdx.x * blockDim.x + threadIdx.x;
  if (i >= rows * Ncols) return;
  const int r = i / Ncols, c = i % Ncols;
  const float* x = X + (size_t)r * Kin;
  float acc = b[c];
  for (int k = 0; k < Kin; ++k) acc += x[k] * W[k * Ncols + c];
  Y[i] = acc;
}

__global__ void zero_kernel(float* __restrict__ p, int n) {
  const int i = blockIdx.x * blockDim.x + threadIdx.x;
  if (i < n) p[i] = 0.f;
}

// ---------------- edge MLP: gather + (160->64 tanh) + (64->64), pre-BN out
__global__ __launch_bounds__(32)
void edge_mlp_kernel(const float* __restrict__ atom, const float* __restrict__ state,
                     const float* __restrict__ nbr, const int* __restrict__ idx,
                     const half_t* __restrict__ w1s, const float* __restrict__ b1,
                     const half_t* __restrict__ w2s, const float* __restrict__ b2,
                     float* __restrict__ hbuf, float* __restrict__ s1,
                     float* __restrict__ ss1) {
  __shared__ half_t ein[16 * 160];
  __shared__ half_t h1[16 * 64];
  const int lane = threadIdx.x;
  const int row0 = blockIdx.x * 16;
  {
    const int r = lane >> 1, sub = lane & 1;
    const int g = row0 + r;
    const int n = g / M_NBR;
    const int nb = idx[g];
    half_t* dst = ein + r * 160;
    if (sub == 0) {
      const float* pa = atom  + (size_t)n * 32;
      const float* ps = state + (size_t)n * 32;
      const float* pn = nbr   + (size_t)g * 32;
#pragma unroll
      for (int o = 0; o < 32; ++o) {
        dst[o]      = (half_t)pa[o];
        dst[32 + o] = (half_t)ps[o];
        dst[64 + o] = (half_t)pn[o];
      }
    } else {
      const float* pa = atom  + (size_t)nb * 32;
      const float* ps = state + (size_t)nb * 32;
#pragma unroll
      for (int o = 0; o < 32; ++o) {
        dst[96 + o]  = (half_t)pa[o];
        dst[128 + o] = (half_t)ps[o];
      }
    }
  }
  __syncthreads();

  v8f acc[4];
#pragma unroll
  for (int t = 0; t < 4; ++t) acc[t] = V8F_ZERO;
#pragma unroll
  for (int c = 0; c < 5; ++c) {
    const v16h a = load_a_frag(ein, 160, c * 32, lane);
#pragma unroll
    for (int t = 0; t < 4; ++t)
      acc[t] = wmma_f16(a, load_b_frag(w1s, c, 4, t, lane), acc[t]);
  }

  const int col16 = lane & 15, rowoff = (lane >> 4) << 3;
#pragma unroll
  for (int t = 0; t < 4; ++t) {
    const int ncol = t * 16 + col16;
    const float bb = b1[ncol];
#pragma unroll
    for (int rr = 0; rr < 8; ++rr)
      h1[(rowoff + rr) * 64 + ncol] = (half_t)tanhf(acc[t][rr] + bb);
  }
  __syncthreads();

#pragma unroll
  for (int t = 0; t < 4; ++t) acc[t] = V8F_ZERO;
#pragma unroll
  for (int c = 0; c < 2; ++c) {
    const v16h a = load_a_frag(h1, 64, c * 32, lane);
#pragma unroll
    for (int t = 0; t < 4; ++t)
      acc[t] = wmma_f16(a, load_b_frag(w2s, c, 4, t, lane), acc[t]);
  }

#pragma unroll
  for (int t = 0; t < 4; ++t) {
    const int ncol = t * 16 + col16;
    const float bb = b2[ncol];
    float sm = 0.f, sq = 0.f;
#pragma unroll
    for (int rr = 0; rr < 8; ++rr) {
      const float v = acc[t][rr] + bb;
      hbuf[(size_t)(row0 + rowoff + rr) * 64 + ncol] = v;
      sm += v; sq += v * v;
    }
    sm += __shfl_xor(sm, 16);
    sq += __shfl_xor(sq, 16);
    if (lane < 16) { atomicAdd(&s1[ncol], sm); atomicAdd(&ss1[ncol], sq); }
  }
}

__global__ void edge_apply_kernel(const float* __restrict__ hbuf,
                                  const float* __restrict__ s, const float* __restrict__ ss,
                                  const float* __restrict__ g, const float* __restrict__ b,
                                  float* __restrict__ nbr) {
  const int i = blockIdx.x * blockDim.x + threadIdx.x;
  if (i >= NM * 32) return;
  const int gi = i >> 5, j = i & 31;
  const float invc = 1.0f / (float)NM;
  const float filt = bn_val(hbuf[(size_t)gi * 64 + j],      s, ss, g, b, j,      invc);
  const float core = bn_val(hbuf[(size_t)gi * 64 + j + 32], s, ss, g, b, j + 32, invc);
  const size_t o = (size_t)gi * 32 + j;
  nbr[o] = filt * core + nbr[o];
}

__global__ void edge_mean_kernel(const float* __restrict__ nbr, float* __restrict__ emean) {
  const int i = blockIdx.x * blockDim.x + threadIdx.x;
  if (i >= N_ATOMS * 32) return;
  const int n = i >> 5, j = i & 31;
  float acc = 0.f;
  for (int m = 0; m < M_NBR; ++m) acc += nbr[(size_t)(n * M_NBR + m) * 32 + j];
  emean[i] = acc * (1.0f / M_NBR);
}

// ---------------- vertex MLP: [emean|atom] -> tanh(64) -> tanh(64), pre-BN out
__global__ __launch_bounds__(32)
void vertex_mlp_kernel(const float* __restrict__ emean, const float* __restrict__ atom,
                       const half_t* __restrict__ w1s, const float* __restrict__ b1,
                       const half_t* __restrict__ w2s, const float* __restrict__ b2,
                       float* __restrict__ hvbuf, float* __restrict__ s2,
                       float* __restrict__ ss2) {
  __shared__ half_t ein[16 * 64];
  __shared__ half_t h1[16 * 64];
  const int lane = threadIdx.x;
  const int row0 = blockIdx.x * 16;
  {
    const int r = lane >> 1, sub = lane & 1;
    const size_t n32 = (size_t)(row0 + r) * 32;
    const float* src = sub ? (atom + n32) : (emean + n32);
    half_t* dst = ein + r * 64 + sub * 32;
#pragma unroll
    for (int o = 0; o < 32; ++o) dst[o] = (half_t)src[o];
  }
  __syncthreads();

  v8f acc[4];
#pragma unroll
  for (int t = 0; t < 4; ++t) acc[t] = V8F_ZERO;
#pragma unroll
  for (int c = 0; c < 2; ++c) {
    const v16h a = load_a_frag(ein, 64, c * 32, lane);
#pragma unroll
    for (int t = 0; t < 4; ++t)
      acc[t] = wmma_f16(a, load_b_frag(w1s, c, 4, t, lane), acc[t]);
  }

  const int col16 = lane & 15, rowoff = (lane >> 4) << 3;
#pragma unroll
  for (int t = 0; t < 4; ++t) {
    const int ncol = t * 16 + col16;
    const float bb = b1[ncol];
#pragma unroll
    for (int rr = 0; rr < 8; ++rr)
      h1[(rowoff + rr) * 64 + ncol] = (half_t)tanhf(acc[t][rr] + bb);
  }
  __syncthreads();

#pragma unroll
  for (int t = 0; t < 4; ++t) acc[t] = V8F_ZERO;
#pragma unroll
  for (int c = 0; c < 2; ++c) {
    const v16h a = load_a_frag(h1, 64, c * 32, lane);
#pragma unroll
    for (int t = 0; t < 4; ++t)
      acc[t] = wmma_f16(a, load_b_frag(w2s, c, 4, t, lane), acc[t]);
  }

#pragma unroll
  for (int t = 0; t < 4; ++t) {
    const int ncol = t * 16 + col16;
    const float bb = b2[ncol];
    float sm = 0.f, sq = 0.f;
#pragma unroll
    for (int rr = 0; rr < 8; ++rr) {
      const float v = tanhf(acc[t][rr] + bb);
      hvbuf[(size_t)(row0 + rowoff + rr) * 64 + ncol] = v;
      sm += v; sq += v * v;
    }
    sm += __shfl_xor(sm, 16);
    sq += __shfl_xor(sq, 16);
    if (lane < 16) { atomicAdd(&s2[ncol], sm); atomicAdd(&ss2[ncol], sq); }
  }
}

__global__ void vertex_apply_kernel(const float* __restrict__ hvbuf,
                                    const float* __restrict__ s, const float* __restrict__ ss,
                                    const float* __restrict__ g, const float* __restrict__ b,
                                    float* __restrict__ atom) {
  const int i = blockIdx.x * blockDim.x + threadIdx.x;
  if (i >= N_ATOMS * 32) return;
  const int n = i >> 5, j = i & 31;
  const float invc = 1.0f / (float)N_ATOMS;
  const float filt = bn_val(hvbuf[(size_t)n * 64 + j],      s, ss, g, b, j,      invc);
  const float core = bn_val(hvbuf[(size_t)n * 64 + j + 32], s, ss, g, b, j + 32, invc);
  atom[i] = filt * core + atom[i];
}

// ---------------- state: 32x32 GEMM, pre-BN out
__global__ __launch_bounds__(32)
void state_mlp_kernel(const float* __restrict__ state, const half_t* __restrict__ wus,
                      const float* __restrict__ bu, float* __restrict__ sbuf,
                      float* __restrict__ s3, float* __restrict__ ss3) {
  __shared__ half_t ein[16 * 32];
  const int lane = threadIdx.x;
  const int row0 = blockIdx.x * 16;
  {
    const int r = lane >> 1, sub = lane & 1;
    const float* p = state + (size_t)(row0 + r) * 32 + sub * 16;
    half_t* dst = ein + r * 32 + sub * 16;
#pragma unroll
    for (int o = 0; o < 16; ++o) dst[o] = (half_t)p[o];
  }
  __syncthreads();

  v8f acc[2];
  acc[0] = V8F_ZERO; acc[1] = V8F_ZERO;
  const v16h a = load_a_frag(ein, 32, 0, lane);
#pragma unroll
  for (int t = 0; t < 2; ++t)
    acc[t] = wmma_f16(a, load_b_frag(wus, 0, 2, t, lane), acc[t]);

  const int col16 = lane & 15, rowoff = (lane >> 4) << 3;
#pragma unroll
  for (int t = 0; t < 2; ++t) {
    const int ncol = t * 16 + col16;
    const float bb = bu[ncol];
    float sm = 0.f, sq = 0.f;
#pragma unroll
    for (int rr = 0; rr < 8; ++rr) {
      const float v = acc[t][rr] + bb;
      sbuf[(size_t)(row0 + rowoff + rr) * 32 + ncol] = v;
      sm += v; sq += v * v;
    }
    sm += __shfl_xor(sm, 16);
    sq += __shfl_xor(sq, 16);
    if (lane < 16) { atomicAdd(&s3[ncol], sm); atomicAdd(&ss3[ncol], sq); }
  }
}

__global__ void state_apply_kernel(const float* __restrict__ sbuf,
                                   const float* __restrict__ s, const float* __restrict__ ss,
                                   const float* __restrict__ g, const float* __restrict__ b,
                                   float* __restrict__ state) {
  const int i = blockIdx.x * blockDim.x + threadIdx.x;
  if (i >= N_ATOMS * 32) return;
  const int j = i & 31;
  const float invc = 1.0f / (float)N_ATOMS;
  state[i] = bn_val(sbuf[i], s, ss, g, b, j, invc) + state[i];
}

// ---------------- readout ----------------
__global__ void nbr_sum_kernel(const float* __restrict__ nbr, float* __restrict__ outp) {
  const int i = blockIdx.x * blockDim.x + threadIdx.x;
  if (i >= N_ATOMS * 32) return;
  const int n = i >> 5, j = i & 31;
  float acc = 0.f;
  for (int m = 0; m < M_NBR; ++m) acc += nbr[(size_t)(n * M_NBR + m) * 32 + j];
  outp[i] = acc;
}

__global__ void pool_kernel(const float* __restrict__ nbrsum, const float* __restrict__ atom,
                            const float* __restrict__ state, const int* __restrict__ cidx,
                            float* __restrict__ crys) {
  const int i = blockIdx.x * blockDim.x + threadIdx.x;
  if (i >= B_CRYS * 96) return;
  const int b = i / 96, f = i % 96;
  const float* src = (f < 32) ? nbrsum : ((f < 64) ? atom : state);
  const int fo = f & 31;
  float acc = 0.f;
  for (int a = 0; a < NPC; ++a) {
    const int n = cidx[b * NPC + a];
    acc += src[(size_t)n * 32 + fo];
  }
  crys[(size_t)b * 108 + f] = acc * (1.0f / NPC);
}

// ---------------- lenet: one workgroup per image, all stages in LDS
__global__ __launch_bounds__(128)
void lenet_kernel(const float* __restrict__ surf,
                  const float* __restrict__ c1w, const float* __restrict__ c1b,
                  const float* __restrict__ c2w, const float* __restrict__ c2b,
                  const float* __restrict__ fc3w, const float* __restrict__ fc3b,
                  float* __restrict__ crys) {
  __shared__ float img[27 * 27];
  __shared__ float p1[3 * 13 * 13];
  __shared__ float p2[6 * 6 * 6];
  const int b = blockIdx.x, t = threadIdx.x;
  const float* im = surf + (size_t)b * 729;   // channel 0 only (x[:, :1])
  for (int i = t; i < 729; i += 128) img[i] = im[i];
  __syncthreads();
  // conv1 (3,1,2,2) VALID -> 26x26, maxpool2 -> 13x13, relu
  for (int i = t; i < 3 * 169; i += 128) {
    const int c = i / 169, rem = i % 169, y = rem / 13, x = rem % 13;
    const float w00 = c1w[c * 4], w01 = c1w[c * 4 + 1], w10 = c1w[c * 4 + 2], w11 = c1w[c * 4 + 3];
    const float bb = c1b[c];
    float mx = -INFINITY;
#pragma unroll
    for (int dy = 0; dy < 2; ++dy)
#pragma unroll
      for (int dx = 0; dx < 2; ++dx) {
        const int yy = 2 * y + dy, xx = 2 * x + dx;
        const float v = img[yy * 27 + xx] * w00 + img[yy * 27 + xx + 1] * w01 +
                        img[(yy + 1) * 27 + xx] * w10 + img[(yy + 1) * 27 + xx + 1] * w11 + bb;
        mx = fmaxf(mx, v);
      }
    p1[i] = fmaxf(mx, 0.0f);
  }
  __syncthreads();
  // conv2 (6,3,2,2) -> 6x12x12, maxpool2 -> 6x6x6, relu
  for (int i = t; i < 6 * 36; i += 128) {
    const int c = i / 36, rem = i % 36, y = rem / 6, x = rem % 6;
    const float bb = c2b[c];
    float mx = -INFINITY;
#pragma unroll
    for (int dy = 0; dy < 2; ++dy)
#pragma unroll
      for (int dx = 0; dx < 2; ++dx) {
        const int yy = 2 * y + dy, xx = 2 * x + dx;
        float v = bb;
#pragma unroll
        for (int ic = 0; ic < 3; ++ic) {
          const float* wp = c2w + (c * 3 + ic) * 4;
          const float* ip = p1 + ic * 169 + yy * 13 + xx;
          v += ip[0] * wp[0] + ip[1] * wp[1] + ip[13] * wp[2] + ip[14] * wp[3];
        }
        mx = fmaxf(mx, v);
      }
    p2[i] = fmaxf(mx, 0.0f);
  }
  __syncthreads();
  // fc3: 216 -> 12, tanh; append to crys cols 96..107
  if (t < 12) {
    float acc = fc3b[t];
    for (int k = 0; k < 216; ++k) acc += p2[k] * fc3w[k * 12 + t];
    crys[(size_t)b * 108 + 96 + t] = tanhf(acc);
  }
}

// ---------------- head: tanh(crys @ fc1) @ out_w + out_b
__global__ __launch_bounds__(128)
void head_kernel(const float* __restrict__ crys, const float* __restrict__ fc1w,
                 const float* __restrict__ fc1b, const float* __restrict__ outw,
                 const float* __restrict__ outb, float* __restrict__ out) {
  __shared__ float xrow[108];
  __shared__ float hred[128];
  const int b = blockIdx.x, t = threadIdx.x;
  if (t < 108) xrow[t] = crys[(size_t)b * 108 + t];
  __syncthreads();
  float acc = fc1b[t];
  for (int k = 0; k < 108; ++k) acc += xrow[k] * fc1w[k * 128 + t];
  hred[t] = tanhf(acc) * outw[t];
  __syncthreads();
  for (int sgap = 64; sgap > 0; sgap >>= 1) {
    if (t < sgap) hred[t] += hred[t + sgap];
    __syncthreads();
  }
  if (t == 0) out[b] = hred[0] + outb[0];
}

// ===========================================================================
extern "C" void kernel_launch(void* const* d_in, const int* in_sizes, int n_in,
                              void* d_out, int out_size, void* d_ws, size_t ws_size,
                              hipStream_t stream) {
  if (n_in < 70) return;
  // Inputs: setup_inputs() insertion order, params flattened recursively.
  const float* atom_fea  = (const float*)d_in[0];
  const float* nbr_fea   = (const float*)d_in[1];
  const float* state_fea = (const float*)d_in[2];
  const float* surface   = (const float*)d_in[3];
  const float* emb_a_w = (const float*)d_in[4];
  const float* emb_a_b = (const float*)d_in[5];
  const float* emb_s_w = (const float*)d_in[6];
  const float* emb_s_b = (const float*)d_in[7];
  const float* emb_n_w = (const float*)d_in[8];
  const float* emb_n_b = (const float*)d_in[9];
  const int BLK0 = 10;     // 16 leaves per block
  const float* c1w  = (const float*)d_in[58];
  const float* c1b  = (const float*)d_in[59];
  const float* c2w  = (const float*)d_in[60];
  const float* c2b  = (const float*)d_in[61];
  const float* fc3w = (const float*)d_in[62];
  const float* fc3b = (const float*)d_in[63];
  const float* fc1w = (const float*)d_in[64];
  const float* fc1b = (const float*)d_in[65];
  const float* outw = (const float*)d_in[66];
  const float* outb = (const float*)d_in[67];
  const int* nbr_idx = (const int*)d_in[68];
  const int* cidx    = (const int*)d_in[69];
  float* out = (float*)d_out;

  // ---- workspace carve (aligned to 256B) ----
  char* wsb = (char*)d_ws;
  size_t off = 0;
  auto carve = [&](size_t bytes) -> void* {
    void* p = wsb + off;
    off = (off + bytes + 255) & ~(size_t)255;
    return p;
  };
  half_t* wswz  = (half_t*)carve((size_t)NBLKS * BLK_WSTRIDE * sizeof(half_t));
  float* atomb  = (float*)carve((size_t)N_ATOMS * 32 * 4);
  float* stateb = (float*)carve((size_t)N_ATOMS * 32 * 4);
  float* nbrb   = (float*)carve((size_t)NM * 32 * 4);
  float* hbuf   = (float*)carve((size_t)NM * 64 * 4);
  float* hvbuf  = (float*)carve((size_t)N_ATOMS * 64 * 4);
  float* sbuf   = (float*)carve((size_t)N_ATOMS * 32 * 4);
  float* emean  = (float*)carve((size_t)N_ATOMS * 32 * 4);
  float* nbrsum = (float*)carve((size_t)N_ATOMS * 32 * 4);
  float* crys   = (float*)carve((size_t)B_CRYS * 108 * 4);
  float* stats  = (float*)carve(320 * 4);
  float* s1 = stats, *ss1 = stats + 64;
  float* s2 = stats + 128, *ss2 = stats + 192;
  float* s3 = stats + 256, *ss3 = stats + 288;

  // ---- stage block weights as f16 WMMA B-fragments ----
  for (int k = 0; k < NBLKS; ++k) {
    const float* e1w = (const float*)d_in[BLK0 + 16 * k + 0];
    const float* e2w = (const float*)d_in[BLK0 + 16 * k + 2];
    const float* v1w = (const float*)d_in[BLK0 + 16 * k + 4];
    const float* v2w = (const float*)d_in[BLK0 + 16 * k + 6];
    const float* uw  = (const float*)d_in[BLK0 + 16 * k + 8];
    half_t* base = wswz + (size_t)k * BLK_WSTRIDE;
    swizzle_w_kernel<<<40, 256, 0, stream>>>(e1w, base + E1_OFF, 160, 64);
    swizzle_w_kernel<<<16, 256, 0, stream>>>(e2w, base + E2_OFF, 64, 64);
    swizzle_w_kernel<<<16, 256, 0, stream>>>(v1w, base + V1_OFF, 64, 64);
    swizzle_w_kernel<<<16, 256, 0, stream>>>(v2w, base + V2_OFF, 64, 64);
    swizzle_w_kernel<<<4, 256, 0, stream>>>(uw, base + U_OFF, 32, 32);
  }

  // ---- embeddings ----
  emb_kernel<<<(N_ATOMS * 32 + 255) / 256, 256, 0, stream>>>(atom_fea, emb_a_w, emb_a_b,
                                                             atomb, N_ATOMS, 92, 32);
  emb_kernel<<<(N_ATOMS * 32 + 255) / 256, 256, 0, stream>>>(state_fea, emb_s_w, emb_s_b,
                                                             stateb, N_ATOMS, 41, 32);
  emb_kernel<<<(NM * 32 + 255) / 256, 256, 0, stream>>>(nbr_fea, emb_n_w, emb_n_b,
                                                        nbrb, NM, 41, 32);

  // ---- cluster blocks ----
  for (int k = 0; k < NBLKS; ++k) {
    const float* e1b = (const float*)d_in[BLK0 + 16 * k + 1];
    const float* e2b = (const float*)d_in[BLK0 + 16 * k + 3];
    const float* v1b = (const float*)d_in[BLK0 + 16 * k + 5];
    const float* v2b = (const float*)d_in[BLK0 + 16 * k + 7];
    const float* ub  = (const float*)d_in[BLK0 + 16 * k + 9];
    const float* bn1g = (const float*)d_in[BLK0 + 16 * k + 10];
    const float* bn1b = (const float*)d_in[BLK0 + 16 * k + 11];
    const float* bn2g = (const float*)d_in[BLK0 + 16 * k + 12];
    const float* bn2b = (const float*)d_in[BLK0 + 16 * k + 13];
    const float* bn3g = (const float*)d_in[BLK0 + 16 * k + 14];
    const float* bn3b = (const float*)d_in[BLK0 + 16 * k + 15];
    half_t* base = wswz + (size_t)k * BLK_WSTRIDE;

    zero_kernel<<<2, 256, 0, stream>>>(stats, 320);
    edge_mlp_kernel<<<NM / 16, 32, 0, stream>>>(atomb, stateb, nbrb, nbr_idx,
                                                base + E1_OFF, e1b, base + E2_OFF, e2b,
                                                hbuf, s1, ss1);
    edge_apply_kernel<<<(NM * 32 + 255) / 256, 256, 0, stream>>>(hbuf, s1, ss1, bn1g, bn1b, nbrb);
    edge_mean_kernel<<<(N_ATOMS * 32 + 255) / 256, 256, 0, stream>>>(nbrb, emean);
    vertex_mlp_kernel<<<N_ATOMS / 16, 32, 0, stream>>>(emean, atomb,
                                                       base + V1_OFF, v1b, base + V2_OFF, v2b,
                                                       hvbuf, s2, ss2);
    vertex_apply_kernel<<<(N_ATOMS * 32 + 255) / 256, 256, 0, stream>>>(hvbuf, s2, ss2,
                                                                        bn2g, bn2b, atomb);
    state_mlp_kernel<<<N_ATOMS / 16, 32, 0, stream>>>(stateb, base + U_OFF, ub, sbuf, s3, ss3);
    state_apply_kernel<<<(N_ATOMS * 32 + 255) / 256, 256, 0, stream>>>(sbuf, s3, ss3,
                                                                       bn3g, bn3b, stateb);
  }

  // ---- readout ----
  nbr_sum_kernel<<<(N_ATOMS * 32 + 255) / 256, 256, 0, stream>>>(nbrb, nbrsum);
  pool_kernel<<<(B_CRYS * 96 + 255) / 256, 256, 0, stream>>>(nbrsum, atomb, stateb, cidx, crys);
  lenet_kernel<<<B_CRYS, 128, 0, stream>>>(surface, c1w, c1b, c2w, c2b, fc3w, fc3b, crys);
  head_kernel<<<B_CRYS, 128, 0, stream>>>(crys, fc1w, fc1b, outw, outb, out);
}
